// HTR_28123445854680
// MI455X (gfx1250) — compile-verified, matrix operand order checked by reading, not run on hardware
//
#include <hip/hip_runtime.h>
#include <hip/hip_bf16.h>

// Problem constants (match reference)
#define NN 8192      // nodes
#define NE 131072    // edges
#define DD 128       // feature dim
#define RR 32        // rank

typedef _Float16 half8  __attribute__((ext_vector_type(8)));
typedef _Float16 half16 __attribute__((ext_vector_type(16)));
typedef float    float8 __attribute__((ext_vector_type(8)));
typedef float    float4v __attribute__((ext_vector_type(4)));

// ---------------------------------------------------------------------------
// WMMA helpers (v_wmma_f32_16x16x32_f16, wave32)
// A layout (16x32 f16): lane L<16 -> row L, K runs {0..7},{16..23};
//                       lane L>=16 -> row L-16, K runs {8..15},{24..31}.
// B layout (32x16 f16): lane = column, same K striping (stored pre-swizzled).
// C/D layout (16x16 f32): VGPR v -> row v + (lane>=16 ? 8 : 0), col = lane&15.
// ---------------------------------------------------------------------------
static __device__ __forceinline__ float8 wmma16(half16 a, half16 b, float8 c) {
  return __builtin_amdgcn_wmma_f32_16x16x32_f16(false, a, false, b, (short)0, c,
                                                false, false);
}

// Build A operand from fp32 row data in global memory. rowPtr = row base + k0.
static __device__ __forceinline__ half16 a_from_f32(const float* rowPtr, int lane) {
  const float* p = rowPtr + ((lane & 16) ? 8 : 0);
  float4v x0 = *(const float4v*)(p + 0);
  float4v x1 = *(const float4v*)(p + 4);
  float4v x2 = *(const float4v*)(p + 16);
  float4v x3 = *(const float4v*)(p + 20);
  half16 a;
#pragma unroll
  for (int j = 0; j < 4; ++j) {
    a[j]      = (_Float16)x0[j];
    a[4 + j]  = (_Float16)x1[j];
    a[8 + j]  = (_Float16)x2[j];
    a[12 + j] = (_Float16)x3[j];
  }
  return a;
}

// Build A operand from f16 LDS tile with row stride `ldRow` halves.
static __device__ __forceinline__ half16 a_from_lds(const _Float16* tile, int ldRow,
                                                    int k0, int lane) {
  const _Float16* p = tile + (lane & 15) * ldRow + k0 + ((lane & 16) ? 8 : 0);
  half8 lo = *(const half8*)(p);
  half8 hi = *(const half8*)(p + 16);
  half16 a;
#pragma unroll
  for (int j = 0; j < 8; ++j) { a[j] = lo[j]; a[8 + j] = hi[j]; }
  return a;
}

// Load pre-swizzled B tile: 32 lanes x 16 halves, contiguous per lane.
static __device__ __forceinline__ half16 b_from_sw(const _Float16* sw, int tile,
                                                   int lane) {
  return *(const half16*)(sw + ((size_t)tile * 32 + lane) * 16);
}

// ---------------------------------------------------------------------------
// Kernel 1: convert fp32 weight [K x C] into f16 WMMA-B swizzled layout.
// Tile index = kb * (C/16) + nb ; element t = tile*512 + lane*16 + j.
// ---------------------------------------------------------------------------
__global__ void k_swizzle(const float* __restrict__ W, _Float16* __restrict__ sw,
                          int K, int C) {
  int t = blockIdx.x * blockDim.x + threadIdx.x;
  if (t >= K * C) return;
  int j    = t & 15;
  int lane = (t >> 4) & 31;
  int tile = t >> 9;
  int nNB  = C >> 4;
  int kb   = tile / nNB;
  int nb   = tile - kb * nNB;
  int col  = nb * 16 + (lane & 15);
  int kk   = kb * 32 + ((lane & 16) ? 8 : 0) + (j & 7) + ((j & 8) ? 16 : 0);
  sw[t] = (_Float16)W[kk * C + col];
}

// ---------------------------------------------------------------------------
// Kernel 2: node projection GEMM  [nRows x 128] @ [128 x 32]
// One wave per 16-row tile. Output layout is channel-major [n][32][16]:
// slot = degOff + l (slots 0..14 used, 15 is pad) so the edge kernel can
// gather one lane's 15 l-values with contiguous b128 loads.
// ---------------------------------------------------------------------------
__global__ __launch_bounds__(256) void k_proj(const float* __restrict__ X,
                                              const _Float16* __restrict__ Wsw,
                                              float* __restrict__ outBase,
                                              int s, int degOff, int nRows) {
  int gwave = (blockIdx.x * blockDim.x + threadIdx.x) >> 5;
  int lane  = threadIdx.x & 31;
  int rowBase = gwave * 16;
  if (rowBase >= nRows) return;

  float8 acc[2] = {};
  const float* aRow = X + (size_t)(rowBase + (lane & 15)) * DD;
#pragma unroll
  for (int ks = 0; ks < 4; ++ks) {
    half16 a = a_from_f32(aRow + ks * 32, lane);
#pragma unroll
    for (int ct = 0; ct < 2; ++ct)
      acc[ct] = wmma16(a, b_from_sw(Wsw, ks * 2 + ct, lane), acc[ct]);
  }

  int colL = lane & 15;
  int rAdd = (lane & 16) ? 8 : 0;
#pragma unroll
  for (int ct = 0; ct < 2; ++ct)
#pragma unroll
    for (int v = 0; v < 8; ++v) {
      int r  = rowBase + v + rAdd;
      int n  = r / s;
      int l  = r - n * s;
      int ch = ct * 16 + colL;
      outBase[((size_t)n * RR + ch) * 16 + degOff + l] = acc[ct][v];
    }
}

// ---------------------------------------------------------------------------
// Kernel 3: fused edge kernel. One wave handles 16 edges.
// ---------------------------------------------------------------------------
#define EW 4  // waves per block

static __device__ __forceinline__ float silu(float x) {
  return x / (1.f + __expf(-x));
}

__global__ __launch_bounds__(EW * 32) void k_edge(
    const float* __restrict__ t_ij,
    const float* __restrict__ r0, const float* __restrict__ r1,
    const float* __restrict__ r2, const int* __restrict__ edge_index,
    const float* __restrict__ eq, const float* __restrict__ ek,
    const float* __restrict__ ln_g, const float* __restrict__ ln_b,
    const _Float16* __restrict__ Ww_sw, const float* __restrict__ b_w,
    const _Float16* __restrict__ Wt1_sw, const float* __restrict__ b_t1,
    const _Float16* __restrict__ Wt2_sw, const float* __restrict__ b_t2,
    float* __restrict__ out) {
  __shared__ _Float16 lds_wn[EW][16][32];
  __shared__ _Float16 lds_h[EW][16][128];

  int w    = threadIdx.x >> 5;
  int lane = threadIdx.x & 31;
  int e0   = (blockIdx.x * EW + w) * 16;

  // Warm L2/L0 with this wave's t_ij rows while we do the scalar pass.
  {
    const float* tp = t_ij + (size_t)(e0 + (lane & 15)) * DD;
    __builtin_prefetch(tp, 0, 3);
    __builtin_prefetch(tp + 64, 0, 3);
  }

  float g  = ln_g[lane];
  float bb = ln_b[lane];

  // ---- per-edge pass: gather (b128), vector-rejection, dot, LayerNorm ----
  for (int el = 0; el < 16; ++el) {
    int e  = e0 + el;
    int nj = edge_index[e];
    int ni = edge_index[NE + e];
    // lane = channel; 15 l-values contiguous (slot 15 = pad, never used).
    const float4v* eqp = (const float4v*)(eq + (((size_t)ni * RR + lane) << 4));
    const float4v* ekp = (const float4v*)(ek + (((size_t)nj * RR + lane) << 4));
    float q[16], kk[16];
#pragma unroll
    for (int g2 = 0; g2 < 4; ++g2) {
      float4v vq = eqp[g2];
      float4v vk = ekp[g2];
#pragma unroll
      for (int j = 0; j < 4; ++j) {
        q[g2 * 4 + j]  = vq[j];
        kk[g2 * 4 + j] = vk[j];
      }
    }
    float wsum = 0.f;
    // degree 0 (s=3, slots 0..2) — note _vec_rej(x,-r) == _vec_rej(x,r)
    {
      float a[3];
#pragma unroll
      for (int c = 0; c < 3; ++c) a[c] = r0[e * 3 + c];
      float dq = 0.f, dk = 0.f;
#pragma unroll
      for (int c = 0; c < 3; ++c) { dq += q[c] * a[c]; dk += kk[c] * a[c]; }
#pragma unroll
      for (int c = 0; c < 3; ++c)
        wsum += (q[c] - dq * a[c]) * (kk[c] - dk * a[c]);
    }
    // degree 1 (s=5, slots 3..7)
    {
      float a[5];
#pragma unroll
      for (int c = 0; c < 5; ++c) a[c] = r1[e * 5 + c];
      float dq = 0.f, dk = 0.f;
#pragma unroll
      for (int c = 0; c < 5; ++c) { dq += q[3 + c] * a[c]; dk += kk[3 + c] * a[c]; }
#pragma unroll
      for (int c = 0; c < 5; ++c)
        wsum += (q[3 + c] - dq * a[c]) * (kk[3 + c] - dk * a[c]);
    }
    // degree 2 (s=7, slots 8..14)
    {
      float a[7];
#pragma unroll
      for (int c = 0; c < 7; ++c) a[c] = r2[e * 7 + c];
      float dq = 0.f, dk = 0.f;
#pragma unroll
      for (int c = 0; c < 7; ++c) { dq += q[8 + c] * a[c]; dk += kk[8 + c] * a[c]; }
#pragma unroll
      for (int c = 0; c < 7; ++c)
        wsum += (q[8 + c] - dq * a[c]) * (kk[8 + c] - dk * a[c]);
    }
    // LayerNorm across R=32 == across the 32 lanes of this wave.
    float s1 = wsum, s2 = wsum * wsum;
#pragma unroll
    for (int d = 16; d >= 1; d >>= 1) {
      s1 += __shfl_xor(s1, d, 32);
      s2 += __shfl_xor(s2, d, 32);
    }
    float mu  = s1 * (1.f / 32.f);
    float var = s2 * (1.f / 32.f) - mu * mu;
    float wn  = (wsum - mu) * rsqrtf(var + 1e-5f) * g + bb;
    lds_wn[w][el][lane] = (_Float16)wn;
  }

  int colL = lane & 15;
  int rAdd = (lane & 16) ? 8 : 0;

  // ---- GEMM1: h = silu(t @ W_t1 + b_t1), 16x128 tile, 32 WMMAs ----
  {
    float8 acc[8] = {};
    const float* tRow = t_ij + (size_t)(e0 + (lane & 15)) * DD;
#pragma unroll
    for (int ks = 0; ks < 4; ++ks) {
      half16 a = a_from_f32(tRow + ks * 32, lane);
#pragma unroll
      for (int ct = 0; ct < 8; ++ct)
        acc[ct] = wmma16(a, b_from_sw(Wt1_sw, ks * 8 + ct, lane), acc[ct]);
    }
#pragma unroll
    for (int ct = 0; ct < 8; ++ct) {
      float b1 = b_t1[ct * 16 + colL];
#pragma unroll
      for (int v = 0; v < 8; ++v)
        lds_h[w][v + rAdd][ct * 16 + colL] = (_Float16)silu(acc[ct][v] + b1);
    }
  }

  // ---- GEMM2: u = h @ W_t2 + b_t2 ; keep su = silu(u) in registers ----
  float8 su[8] = {};
  {
#pragma unroll
    for (int ks = 0; ks < 4; ++ks) {
      half16 a = a_from_lds(&lds_h[w][0][0], DD, ks * 32, lane);
#pragma unroll
      for (int ct = 0; ct < 8; ++ct)
        su[ct] = wmma16(a, b_from_sw(Wt2_sw, ks * 8 + ct, lane), su[ct]);
    }
#pragma unroll
    for (int ct = 0; ct < 8; ++ct) {
      float b2 = b_t2[ct * 16 + colL];
#pragma unroll
      for (int v = 0; v < 8; ++v) su[ct][v] = silu(su[ct][v] + b2);
    }
  }

  // ---- dw = wn @ W_w + b_w (K=32, one k-step), fused final multiply ----
  {
    half16 a = a_from_lds(&lds_wn[w][0][0], RR, 0, lane);
#pragma unroll
    for (int ct = 0; ct < 8; ++ct) {
      float8 dw = {};
      dw = wmma16(a, b_from_sw(Ww_sw, ct, lane), dw);
      float bw = b_w[ct * 16 + colL];
#pragma unroll
      for (int v = 0; v < 8; ++v)
        out[(size_t)(e0 + v + rAdd) * DD + ct * 16 + colL] =
            (dw[v] + bw) * su[ct][v];
    }
  }
}

// ---------------------------------------------------------------------------
// Host launcher
// ---------------------------------------------------------------------------
extern "C" void kernel_launch(void* const* d_in, const int* in_sizes, int n_in,
                              void* d_out, int out_size, void* d_ws,
                              size_t ws_size, hipStream_t stream) {
  const float* X0   = (const float*)d_in[0];
  const float* X1   = (const float*)d_in[1];
  const float* X2   = (const float*)d_in[2];
  const float* t_ij = (const float*)d_in[3];
  const float* r0   = (const float*)d_in[4];
  const float* r1   = (const float*)d_in[5];
  const float* r2   = (const float*)d_in[6];
  const int*   eidx = (const int*)d_in[7];
  const float* W_vq  = (const float*)d_in[8];
  const float* W_vk0 = (const float*)d_in[9];
  const float* W_vk1 = (const float*)d_in[10];
  const float* W_vk2 = (const float*)d_in[11];
  const float* ln_g  = (const float*)d_in[12];
  const float* ln_b  = (const float*)d_in[13];
  const float* W_w   = (const float*)d_in[14];
  const float* b_w   = (const float*)d_in[15];
  const float* W_t1  = (const float*)d_in[16];
  const float* b_t1  = (const float*)d_in[17];
  const float* W_t2  = (const float*)d_in[18];
  const float* b_t2  = (const float*)d_in[19];
  float* out = (float*)d_out;

  // Workspace layout: f16 swizzled weights, then eq/ek fp32 node buffers
  // in channel-major [n][32][16] layout (16 MB each).
  _Float16* ws     = (_Float16*)d_ws;
  _Float16* Wvq_sw  = ws + 0;
  _Float16* Wvk0_sw = ws + 4096;
  _Float16* Wvk1_sw = ws + 8192;
  _Float16* Wvk2_sw = ws + 12288;
  _Float16* Ww_sw   = ws + 16384;
  _Float16* Wt1_sw  = ws + 20480;
  _Float16* Wt2_sw  = ws + 36864;  // ends at 53248 halves
  float* eqBuf = (float*)((char*)d_ws + (128u * 1024u));
  float* ekBuf = eqBuf + (size_t)NN * RR * 16;

  // 1) weight convert + swizzle
  k_swizzle<<<16, 256, 0, stream>>>(W_vq, Wvq_sw, 128, 32);
  k_swizzle<<<16, 256, 0, stream>>>(W_vk0, Wvk0_sw, 128, 32);
  k_swizzle<<<16, 256, 0, stream>>>(W_vk1, Wvk1_sw, 128, 32);
  k_swizzle<<<16, 256, 0, stream>>>(W_vk2, Wvk2_sw, 128, 32);
  k_swizzle<<<16, 256, 0, stream>>>(W_w, Ww_sw, 32, 128);
  k_swizzle<<<64, 256, 0, stream>>>(W_t1, Wt1_sw, 128, 128);
  k_swizzle<<<64, 256, 0, stream>>>(W_t2, Wt2_sw, 128, 128);

  // 2) node projections: eq (W_vq on every degree) and ek (per-degree W_vk)
  //    rows per launch: N*s; 16-row tiles, 8 waves (256 threads) per block.
  k_proj<<<(NN * 3 / 16) / 8, 256, 0, stream>>>(X0, Wvq_sw, eqBuf, 3, 0, NN * 3);
  k_proj<<<(NN * 5 / 16) / 8, 256, 0, stream>>>(X1, Wvq_sw, eqBuf, 5, 3, NN * 5);
  k_proj<<<(NN * 7 / 16) / 8, 256, 0, stream>>>(X2, Wvq_sw, eqBuf, 7, 8, NN * 7);
  k_proj<<<(NN * 3 / 16) / 8, 256, 0, stream>>>(X0, Wvk0_sw, ekBuf, 3, 0, NN * 3);
  k_proj<<<(NN * 5 / 16) / 8, 256, 0, stream>>>(X1, Wvk1_sw, ekBuf, 5, 3, NN * 5);
  k_proj<<<(NN * 7 / 16) / 8, 256, 0, stream>>>(X2, Wvk2_sw, ekBuf, 7, 8, NN * 7);

  // 3) fused edge kernel: 16 edges per wave, EW waves per block.
  k_edge<<<NE / (16 * EW), EW * 32, 0, stream>>>(
      t_ij, r0, r1, r2, eidx, eqBuf, ekBuf, ln_g, ln_b, Ww_sw, b_w, Wt1_sw,
      b_t1, Wt2_sw, b_t2, out);
}
